// Message_74835510165535
// MI455X (gfx1250) — matrix operand order repaired and checked by reading, not compiled, not run on hardware
//
#include <hip/hip_runtime.h>
#include <hip/hip_bf16.h>
#include <stdint.h>

typedef __attribute__((ext_vector_type(16))) __bf16 v16bf;
typedef __attribute__((ext_vector_type(8)))  __bf16 v8bf;
typedef __attribute__((ext_vector_type(8)))  float  v8f;

#define N_FEAT   48
#define N_EFEAT  16
#define HIDDEN   128
#define TOT      112      // 2*48 + 16
#define KPAD     128      // zero-padded K for layer 1
#define WAVES    8

// fragment grid: K chunks of 32, N tiles of 16
#define KC1 4
#define NT1 8
#define KC2 4
#define NT2 8
#define KC3 4
#define NT3 3

#define FW1_DW (KC1*NT1*32*8)   // 8192 dwords
#define FW2_DW (KC2*NT2*32*8)   // 8192 dwords
#define FW3_DW (KC3*NT3*32*8)   // 3072 dwords

static __device__ __forceinline__ unsigned pack2_bf16(float a, float b) {
  __bf16 ha = (__bf16)a, hb = (__bf16)b;
  unsigned short ua, ub;
  __builtin_memcpy(&ua, &ha, 2);
  __builtin_memcpy(&ub, &hb, 2);
  return (unsigned)ua | ((unsigned)ub << 16);
}

__global__ void zero_kernel(float* out, int nOut, float* cnt, int nCnt) {
  int i = blockIdx.x * blockDim.x + threadIdx.x;
  if (i < nOut) out[i] = 0.f;
  if (i < nCnt) cnt[i] = 0.f;
}

// Pack row-major f32 weight W[K][N] into WMMA bf16 B-fragment order:
// fragment (kc,nt): 32 lanes x 8 dwords; lane holds col = nt*16+(lane&15),
// K half = (lane>>4)*16; dword j = {W[k0], W[k0+1]} with k0 = kc*32 + half + 2j.
__device__ __forceinline__ void pack_weight(const float* __restrict__ W, int K, int N, int NT,
                                            unsigned* __restrict__ frag, int idx) {
  int j    = idx & 7;
  int lane = (idx >> 3) & 31;
  int f    = idx >> 8;
  int nt   = f % NT;
  int kc   = f / NT;
  int col  = nt * 16 + (lane & 15);
  int k0   = kc * 32 + ((lane >> 4) << 4) + 2 * j;
  float a = (k0     < K) ? W[k0 * N + col]       : 0.f;
  float b = (k0 + 1 < K) ? W[(k0 + 1) * N + col] : 0.f;
  frag[idx] = pack2_bf16(a, b);
}

__global__ void prepack_kernel(const float* __restrict__ W1, const float* __restrict__ W2,
                               const float* __restrict__ W3,
                               unsigned* __restrict__ fw1, unsigned* __restrict__ fw2,
                               unsigned* __restrict__ fw3) {
  int idx = blockIdx.x * blockDim.x + threadIdx.x;
  if (idx < FW1_DW) pack_weight(W1, TOT, HIDDEN, NT1, fw1, idx);
  else if (idx < FW1_DW + FW2_DW) pack_weight(W2, HIDDEN, HIDDEN, NT2, fw2, idx - FW1_DW);
  else if (idx < FW1_DW + FW2_DW + FW3_DW)
    pack_weight(W3, HIDDEN, N_FEAT, NT3, fw3, idx - FW1_DW - FW2_DW);
}

__global__ void count_kernel(const int* __restrict__ eidx, float* __restrict__ cnt, int nEdges) {
  int e = blockIdx.x * blockDim.x + threadIdx.x;
  if (e < nEdges) atomicAdd(&cnt[eidx[nEdges + e]], 1.0f);
}

// A-fragment (16x32 bf16, contiguous k order, same as the B pack): lane holds row = lane&15,
// K offset half = (lane>>4)*16; VGPR0-3 = K [half, half+8), VGPR4-7 = K [half+8, half+16).
__device__ __forceinline__ v16bf load_a_frag(const __bf16* __restrict__ stg, int lane, int kc) {
  int row  = lane & 15;
  int koff = (lane >> 4) << 4;
  const __bf16* p = stg + row * KPAD + kc * 32 + koff;
  v8bf lo = *(const v8bf*)p;          // ds_load_b128
  v8bf hi = *(const v8bf*)(p + 8);    // ds_load_b128
  v16bf a;
  #pragma unroll
  for (int i = 0; i < 8; ++i) { a[i] = lo[i]; a[i + 8] = hi[i]; }
  return a;
}

__device__ __forceinline__ v16bf load_b_frag(const unsigned* __restrict__ fw, int NT,
                                             int kc, int nt, int lane) {
  return *(const v16bf*)(fw + (size_t)(((kc * NT) + nt) * 32 + lane) * 8);  // 2x global_load_b128
}

// Opaque zero offset, kept uniform in an SGPR. Adding it to the (still
// address-space-qualified, global) weight base defeats cross-tile CSE /
// register-caching of the 76 B-fragments (Round-1 spill cause) without
// degrading the loads to flat addressing (Round-2 regression).
static __device__ __forceinline__ int opaque_zero() {
  int z = 0;
  asm volatile("" : "+s"(z));
  return z;
}

__global__ void __launch_bounds__(WAVES * 32) gnn_main(
    const float* __restrict__ x, const int* __restrict__ eidx, const float* __restrict__ ef,
    const unsigned* __restrict__ fw1g, const unsigned* __restrict__ fw2g,
    const unsigned* __restrict__ fw3g,
    const float* __restrict__ b1, const float* __restrict__ b2, const float* __restrict__ b3,
    float* __restrict__ msum, int nEdges) {
  __shared__ __bf16 stageAll[WAVES][16 * KPAD];   // 32 KB: wave-private 16x128 tiles
  __shared__ float  lb[320];                      // b1 | b2 | b3

  int tid  = threadIdx.x;
  int lane = tid & 31;
  int wave = tid >> 5;
  if (tid < HIDDEN) { lb[tid] = b1[tid]; lb[128 + tid] = b2[tid]; }
  if (tid < N_FEAT) lb[256 + tid] = b3[tid];
  __syncthreads();

  __bf16* stage = stageAll[wave];
  int nTiles = (nEdges + 15) >> 4;
  int colb = lane & 15;
  int rowb = (lane >> 4) << 3;

  v8f zero8;
  #pragma unroll
  for (int i = 0; i < 8; ++i) zero8[i] = 0.f;

  for (int tile = blockIdx.x * WAVES + wave; tile < nTiles; tile += gridDim.x * WAVES) {
    int base = tile << 4;

    // per-tile opaque offset: forces per-tile B reloads (L2-resident),
    // keeps global address space -> saddr-form global_load_b128
    int oz = opaque_zero();
    const unsigned* fw1 = fw1g + oz;
    const unsigned* fw2 = fw2g + oz;
    const unsigned* fw3 = fw3g + oz;

    // lanes 0..15 hold this tile's edge endpoints
    int myE = base + (lane & 15);
    bool eok = (lane < 16) && (myE < nEdges);
    int mySrc = eok ? eidx[myE] : 0;
    int myTgt = eok ? eidx[nEdges + myE] : 0;

    // ---- gather + f32->bf16 into LDS staging (16 rows x 128, zero-padded) ----
    // One row per iteration; each lane covers 4 consecutive features via b128 load.
    #pragma unroll 4
    for (int r = 0; r < 16; ++r) {
      int c = lane * 4;
      int e = base + r;
      float4 v = make_float4(0.f, 0.f, 0.f, 0.f);
      if (e < nEdges) {
        int s = __shfl(mySrc, r, 32);
        int t = __shfl(myTgt, r, 32);
        if (c < N_FEAT)          v = *(const float4*)(x + (size_t)s * N_FEAT + c);
        else if (c < 2 * N_FEAT) v = *(const float4*)(x + (size_t)t * N_FEAT + (c - N_FEAT));
        else if (c < TOT)        v = *(const float4*)(ef + (size_t)e * N_EFEAT + (c - 2 * N_FEAT));
      }
      uint2 p;
      p.x = pack2_bf16(v.x, v.y);
      p.y = pack2_bf16(v.z, v.w);
      *(uint2*)(stage + r * KPAD + c) = p;   // ds_store_b64
    }

    // ---- layer 1: (16x128) @ (128x128) ----
    v8f acc[NT1];
    #pragma unroll
    for (int nt = 0; nt < NT1; ++nt) acc[nt] = zero8;
    #pragma unroll
    for (int kc = 0; kc < KC1; ++kc) {
      v16bf a = load_a_frag(stage, lane, kc);
      #pragma unroll
      for (int nt = 0; nt < NT1; ++nt) {
        v16bf b = load_b_frag(fw1, NT1, kc, nt, lane);
        acc[nt] = __builtin_amdgcn_wmma_f32_16x16x32_bf16(false, a, false, b,
                                                          (short)0, acc[nt], false, false);
      }
    }
    #pragma unroll
    for (int nt = 0; nt < NT1; ++nt) {   // bias + relu + restage as bf16
      float bias = lb[nt * 16 + colb];
      #pragma unroll
      for (int v = 0; v < 8; ++v) {
        float r = fmaxf(acc[nt][v] + bias, 0.f);
        stage[(rowb + v) * KPAD + nt * 16 + colb] = (__bf16)r;
      }
    }

    // ---- layer 2: (16x128) @ (128x128) ----
    #pragma unroll
    for (int nt = 0; nt < NT2; ++nt) acc[nt] = zero8;
    #pragma unroll
    for (int kc = 0; kc < KC2; ++kc) {
      v16bf a = load_a_frag(stage, lane, kc);
      #pragma unroll
      for (int nt = 0; nt < NT2; ++nt) {
        v16bf b = load_b_frag(fw2, NT2, kc, nt, lane);
        acc[nt] = __builtin_amdgcn_wmma_f32_16x16x32_bf16(false, a, false, b,
                                                          (short)0, acc[nt], false, false);
      }
    }
    #pragma unroll
    for (int nt = 0; nt < NT2; ++nt) {
      float bias = lb[128 + nt * 16 + colb];
      #pragma unroll
      for (int v = 0; v < 8; ++v) {
        float r = fmaxf(acc[nt][v] + bias, 0.f);
        stage[(rowb + v) * KPAD + nt * 16 + colb] = (__bf16)r;
      }
    }

    // ---- layer 3: (16x128) @ (128x48) + atomic scatter by target ----
    v8f acc3[NT3];
    #pragma unroll
    for (int nt = 0; nt < NT3; ++nt) acc3[nt] = zero8;
    #pragma unroll
    for (int kc = 0; kc < KC3; ++kc) {
      v16bf a = load_a_frag(stage, lane, kc);
      #pragma unroll
      for (int nt = 0; nt < NT3; ++nt) {
        v16bf b = load_b_frag(fw3, NT3, kc, nt, lane);
        acc3[nt] = __builtin_amdgcn_wmma_f32_16x16x32_bf16(false, a, false, b,
                                                           (short)0, acc3[nt], false, false);
      }
    }
    #pragma unroll
    for (int nt = 0; nt < NT3; ++nt) {
      float bias = lb[256 + nt * 16 + colb];
      #pragma unroll
      for (int v = 0; v < 8; ++v) {
        int row = rowb + v;
        int e = base + row;
        if (e < nEdges) {
          int t = __shfl(myTgt, row, 32);
          atomicAdd(&msum[(size_t)t * N_FEAT + nt * 16 + colb], acc3[nt][v] + bias);
        }
      }
    }
  }
}

__global__ void finalize_kernel(const float* __restrict__ x, float* __restrict__ out,
                                const float* __restrict__ cnt, int total) {
  int i = blockIdx.x * blockDim.x + threadIdx.x;
  if (i < total) {
    float c = cnt[i / N_FEAT];
    out[i] = x[i] + out[i] / fmaxf(c, 1.f);
  }
}

extern "C" void kernel_launch(void* const* d_in, const int* in_sizes, int n_in,
                              void* d_out, int out_size, void* d_ws, size_t ws_size,
                              hipStream_t stream) {
  const float* x  = (const float*)d_in[0];
  const int*   ei = (const int*)d_in[1];
  const float* ef = (const float*)d_in[2];
  const float* W1 = (const float*)d_in[3];
  const float* b1 = (const float*)d_in[4];
  const float* W2 = (const float*)d_in[5];
  const float* b2 = (const float*)d_in[6];
  const float* W3 = (const float*)d_in[7];
  const float* b3 = (const float*)d_in[8];
  float* out = (float*)d_out;

  int nNodes = in_sizes[0] / N_FEAT;
  int nEdges = in_sizes[1] / 2;
  int totalOut = nNodes * N_FEAT;

  // workspace: cnt (nNodes f32, 256B aligned) | fw1 | fw2 | fw3
  float* cnt = (float*)d_ws;
  size_t cntBytes = (((size_t)nNodes * sizeof(float)) + 255) & ~(size_t)255;
  unsigned* fw1 = (unsigned*)((char*)d_ws + cntBytes);
  unsigned* fw2 = fw1 + FW1_DW;
  unsigned* fw3 = fw2 + FW2_DW;

  int zn = totalOut > nNodes ? totalOut : nNodes;
  zero_kernel<<<(zn + 255) / 256, 256, 0, stream>>>(out, totalOut, cnt, nNodes);
  prepack_kernel<<<(FW1_DW + FW2_DW + FW3_DW + 255) / 256, 256, 0, stream>>>(W1, W2, W3,
                                                                             fw1, fw2, fw3);
  count_kernel<<<(nEdges + 255) / 256, 256, 0, stream>>>(ei, cnt, nEdges);
  gnn_main<<<1024, WAVES * 32, 0, stream>>>(x, ei, ef, fw1, fw2, fw3, b1, b2, b3, out, nEdges);
  finalize_kernel<<<(totalOut + 255) / 256, 256, 0, stream>>>(x, out, cnt, totalOut);
}
